// TransformerDecoderBlockV2_7241314861848
// MI455X (gfx1250) — matrix operand order repaired and checked by reading, not compiled, not run on hardware
//
#include <hip/hip_runtime.h>
#include <hip/hip_bf16.h>
#include <math.h>

typedef __attribute__((ext_vector_type(16))) __bf16 v16bf;
typedef __attribute__((ext_vector_type(8)))  __bf16 v8bf;
typedef __attribute__((ext_vector_type(8)))  float  v8f;
typedef __attribute__((ext_vector_type(4)))  unsigned int v4u;
typedef __attribute__((ext_vector_type(8)))  int v8i;
typedef __attribute__((ext_vector_type(4)))  int v4i;

#define BM 128
#define BN 128
#define BK 32
#define BKP 40   // padded LDS row stride (elements): 80B, 16B-aligned, bank-spread

#ifndef __has_builtin
#define __has_builtin(x) 0
#endif
#if __has_builtin(__builtin_amdgcn_tensor_load_to_lds)
#define HAVE_TDM 1
#else
#define HAVE_TDM 0
#endif
#if __has_include(<hip/amd_detail/amd_gfx1250_TDM.h>)
#define TDM_SIX_ARG 1
#else
#define TDM_SIX_ARG 0
#endif

#define SHUF16(a, b) __builtin_shufflevector(a, b, 0,1,2,3,4,5,6,7,8,9,10,11,12,13,14,15)

// ---------------------------------------------------------------------------
// TDM: DMA a 2-D bf16 tile (tile_rows x tile_cols) from row-major global into
// LDS at lds_off, inserting 4 DWORDs of pad every 16 DWORDs so 32-elem rows
// land at stride-40 elements. D# built per CDNA5 ISA 8.3/8.4.
// ---------------------------------------------------------------------------
#if HAVE_TDM
__device__ __forceinline__ void tdm_load_2d_bf16(
    unsigned lds_off, const __bf16* gptr, unsigned tile_rows, unsigned tile_cols,
    unsigned row_stride_elems, unsigned tdim0, unsigned tdim1, bool pad)
{
  unsigned long long ga = (unsigned long long)(size_t)gptr;
  v4u g0;
  g0[0] = 1u;                                    // count=1, user descriptor
  g0[1] = lds_off;                               // lds_addr (bytes)
  g0[2] = (unsigned)(ga & 0xFFFFFFFFu);          // global_addr[31:0]
  g0[3] = ((unsigned)(ga >> 32) & 0x01FFFFFFu) | 0x80000000u;  // addr[56:32]|type=2
  unsigned w0 = (1u << 16);                      // data_size = 1 -> 2 bytes
  if (pad) w0 |= (1u << 20) | (3u << 22) | (3u << 25);  // pad_en, every 16DW, +4DW
  v8i g1;
  g1[0] = (int)w0;
  g1[1] = (int)((tdim0 & 0xFFFFu) << 16);                              // tensor_dim0 lo
  g1[2] = (int)(((tdim0 >> 16) & 0xFFFFu) | ((tdim1 & 0xFFFFu) << 16));// dim0 hi, dim1 lo
  g1[3] = (int)(((tdim1 >> 16) & 0xFFFFu) | ((tile_cols & 0xFFFFu) << 16)); // tile_dim0
  g1[4] = (int)(tile_rows & 0xFFFFu);                                  // tile_dim1
  g1[5] = (int)row_stride_elems;                                       // dim0_stride lo
  g1[6] = 0;
  g1[7] = 0;
  v4i z4 = (v4i){0, 0, 0, 0};
#if TDM_SIX_ARG
  v8i z8 = (v8i){0, 0, 0, 0, 0, 0, 0, 0};
  __builtin_amdgcn_tensor_load_to_lds(g0, g1, z4, z4, z8, 0);
#else
  __builtin_amdgcn_tensor_load_to_lds(g0, g1, z4, z4, 0);
#endif
}
#endif

// ---------------------------------------------------------------------------
// f32 -> bf16 convert (grid-stride)
// ---------------------------------------------------------------------------
__global__ void cvt_kernel(const float* __restrict__ in, __bf16* __restrict__ out, int n) {
  int i = blockIdx.x * blockDim.x + threadIdx.x;
  int stride = gridDim.x * blockDim.x;
  for (; i < n; i += stride) out[i] = (__bf16)in[i];
}

// ---------------------------------------------------------------------------
// bf16 WMMA GEMM:  C[M x N] = A[M x K] * B[K x N] + bias  (+ epilogue)
//   EPI 0: store bf16; EPI 1: erf-GELU, store bf16; EPI 2: +residual, store f32
// 256 thr = 8 waves (2x4), wave tile 64x32. Double-buffered LDS: the TDM
// (A tile, padded rows) and the transposed B stage for step k+1 overlap the
// 8 WMMAs of step k; one barrier per K-step flips buffers.
// ---------------------------------------------------------------------------
template<int EPI>
__global__ __launch_bounds__(256)
void gemm_bf16(const __bf16* __restrict__ A, const __bf16* __restrict__ Bm,
               const float* __restrict__ bias, const float* __restrict__ resid,
               float* __restrict__ outF, __bf16* __restrict__ outB,
               int Mdim, int Ndim, int Kdim)
{
  __shared__ __align__(16) __bf16 sA[2][BM * BKP];    // [m][k] padded
  __shared__ __align__(16) __bf16 sBt[2][BN * BKP];   // [n][k] padded (transposed)

  const int tid  = threadIdx.x;
  const int lane = tid & 31;
  const int wave = tid >> 5;
  const int l16  = lane & 15;
  const int hi   = lane >> 4;
  const int wm   = wave >> 2;
  const int wn   = wave & 3;
  const int bm   = blockIdx.y;
  const int bn   = blockIdx.x;

  v8f acc[4][2];
#pragma unroll
  for (int i = 0; i < 4; i++)
#pragma unroll
    for (int j = 0; j < 2; j++) acc[i][j] = (v8f){0.f,0.f,0.f,0.f,0.f,0.f,0.f,0.f};

  // ---- tile stagers ----
  auto stageA = [&](int k0, int buf) {
#if HAVE_TDM
    if (wave == 0)
      tdm_load_2d_bf16((unsigned)(size_t)&sA[buf][0],
                       A + (size_t)(bm * BM) * Kdim + k0,
                       BM, BK, (unsigned)Kdim, (unsigned)Kdim, (unsigned)Mdim, true);
#else
#pragma unroll
    for (int t = 0; t < 2; t++) {
      int idx = (t * 256 + tid) * 8;
      int r = idx >> 5, c = idx & 31;
      *(v8bf*)(&sA[buf][0] + r * BKP + c) =
          *(const v8bf*)(A + (size_t)(bm * BM + r) * Kdim + k0 + c);
    }
#endif
  };
  auto stageB = [&](int k0, int buf) {
#pragma unroll
    for (int t = 0; t < 2; t++) {
      int idx = (t * 256 + tid) * 8;
      int r = idx >> 7, c = idx & 127;   // r = k row, c = n base
      v8bf vv = *(const v8bf*)(Bm + (size_t)(k0 + r) * Ndim + bn * BN + c);
#pragma unroll
      for (int e = 0; e < 8; e++) sBt[buf][(c + e) * BKP + r] = vv[e];
    }
  };

  // ---- prologue: stage tile 0 ----
  stageA(0, 0);
  stageB(0, 0);
#if HAVE_TDM
  if (wave == 0) __builtin_amdgcn_s_wait_tensorcnt(0);
#endif
  __syncthreads();

  int cur = 0;
  for (int k0 = 0; k0 < Kdim; k0 += BK) {
    // ---- async-stage next tile into the other buffer ----
    int nk = k0 + BK;
    if (nk < Kdim) {
      stageA(nk, cur ^ 1);
      stageB(nk, cur ^ 1);
    }

    // ---- compute from current buffer: all-16B vector LDS fragment loads ----
    v16bf af[4];
#pragma unroll
    for (int i = 0; i < 4; i++) {
      const __bf16* ap = &sA[cur][0] + (wm * 64 + i * 16 + l16) * BKP + hi * 8;
      af[i] = SHUF16(*(const v8bf*)ap, *(const v8bf*)(ap + 16));
    }
    v16bf bfr[2];
#pragma unroll
    for (int j2 = 0; j2 < 2; j2++) {
      const __bf16* bp = &sBt[cur][0] + (wn * 32 + j2 * 16 + l16) * BKP + hi * 8;
      bfr[j2] = SHUF16(*(const v8bf*)bp, *(const v8bf*)(bp + 16));
    }
#pragma unroll
    for (int i = 0; i < 4; i++)
#pragma unroll
      for (int j2 = 0; j2 < 2; j2++)
        acc[i][j2] = __builtin_amdgcn_wmma_f32_16x16x32_bf16(
            false, af[i], false, bfr[j2], (short)0, acc[i][j2], false, false);

#if HAVE_TDM
    if (wave == 0) __builtin_amdgcn_s_wait_tensorcnt(0);
#endif
    __syncthreads();
    cur ^= 1;
  }

  // ---- epilogue ----
#pragma unroll
  for (int i = 0; i < 4; i++)
#pragma unroll
    for (int j2 = 0; j2 < 2; j2++)
#pragma unroll
      for (int v = 0; v < 8; v++) {
        int row = bm * BM + wm * 64 + i * 16 + v + 8 * hi;
        int col = bn * BN + wn * 32 + j2 * 16 + l16;
        float val = acc[i][j2][v] + bias[col];
        if (EPI == 1) val = 0.5f * val * (1.0f + erff(val * 0.70710678118654752f));
        if (EPI == 2) {
          val += resid[(size_t)row * Ndim + col];
          outF[(size_t)row * Ndim + col] = val;
        } else {
          outB[(size_t)row * Ndim + col] = (__bf16)val;
        }
      }
}

// ---------------------------------------------------------------------------
// Flash attention. Block = 4 waves (one 16-row tile each) of one head; the
// block shares the per-tile transposed V stage. dh=64, keys=4096.
// ---------------------------------------------------------------------------
__global__ __launch_bounds__(128)
void attn_kernel(const __bf16* __restrict__ Q, const __bf16* __restrict__ KV,
                 __bf16* __restrict__ O)
{
  __shared__ __align__(16) __bf16 sP[4][16 * BKP];  // per-wave P tile
  __shared__ __align__(16) __bf16 sVt[64 * BKP];    // V tile transposed [dim][key]
  const int lane = threadIdx.x & 31;
  const int wave = threadIdx.x >> 5;
  const int l16  = lane & 15;
  const int hi   = lane >> 4;
  const int h    = blockIdx.y;
  const int r0   = blockIdx.x * 64 + wave * 16;
  const float scale = 0.125f;  // 64^-0.5
  const int D = 1024, KVLD = 2048, NM = 4096, Nn = 2048;

  // Q fragments for dh chunks 0..31, 32..63 (two 16B global loads each)
  v16bf aq[2];
#pragma unroll
  for (int c = 0; c < 2; c++) {
    const __bf16* qp = Q + (size_t)(r0 + l16) * D + h * 64 + c * 32 + hi * 8;
    aq[c] = SHUF16(*(const v8bf*)qp, *(const v8bf*)(qp + 16));
  }

  v8f o[4];
#pragma unroll
  for (int f = 0; f < 4; f++) o[f] = (v8f){0.f,0.f,0.f,0.f,0.f,0.f,0.f,0.f};
  float rowm[8], rowl[8];
#pragma unroll
  for (int v = 0; v < 8; v++) { rowm[v] = -1e30f; rowl[v] = 0.f; }

  for (int j0 = 0; j0 < NM; j0 += 32) {
    // ---- cooperative stage: V tile (32 keys x 64 dims) -> sVt[dim][key] ----
#pragma unroll
    for (int t = 0; t < 2; t++) {
      int idx = (t * 128 + (int)threadIdx.x) * 8;
      int r = idx >> 6, c = idx & 63;   // r = key 0..31, c = dim base
      v8bf vv = *(const v8bf*)(KV + (size_t)(j0 + r) * KVLD + 1024 + h * 64 + c);
#pragma unroll
      for (int e = 0; e < 8; e++) sVt[(c + e) * BKP + r] = vv[e];
    }

    // ---- S = (Q*scale) K^T : K fragments are two 16B global loads (k 8h..8h+7, 16+8h..) ----
    v8f s[2];
    s[0] = (v8f){0.f,0.f,0.f,0.f,0.f,0.f,0.f,0.f};
    s[1] = (v8f){0.f,0.f,0.f,0.f,0.f,0.f,0.f,0.f};
#pragma unroll
    for (int t = 0; t < 2; t++)
#pragma unroll
      for (int kc = 0; kc < 2; kc++) {
        const __bf16* kp = KV + (size_t)(j0 + t * 16 + l16) * KVLD + h * 64 + kc * 32 + hi * 8;
        v16bf bk = SHUF16(*(const v8bf*)kp, *(const v8bf*)(kp + 16));
        s[t] = __builtin_amdgcn_wmma_f32_16x16x32_bf16(
            false, aq[kc], false, bk, (short)0, s[t], false, false);
      }

    // ---- mask + online softmax (16-lane half row reductions) ----
#pragma unroll
    for (int v = 0; v < 8; v++) {
      int row = r0 + v + 8 * hi;
      float v0 = s[0][v] * scale, v1 = s[1][v] * scale;
      int c0 = j0 + l16, c1 = j0 + 16 + l16;
      if (c0 < Nn && c0 > row) v0 = -50000.0f;
      if (c1 < Nn && c1 > row) v1 = -50000.0f;
      float mx = fmaxf(v0, v1);
#pragma unroll
      for (int off = 1; off < 16; off <<= 1) mx = fmaxf(mx, __shfl_xor(mx, off));
      float m2 = fmaxf(rowm[v], mx);
      float p0 = __expf(v0 - m2), p1 = __expf(v1 - m2);
      float sum = p0 + p1;
#pragma unroll
      for (int off = 1; off < 16; off <<= 1) sum += __shfl_xor(sum, off);
      float corr = __expf(rowm[v] - m2);
      rowl[v] = rowl[v] * corr + sum;
      rowm[v] = m2;
#pragma unroll
      for (int f = 0; f < 4; f++) o[f][v] *= corr;
      sP[wave][(v + 8 * hi) * BKP + l16]      = (__bf16)p0;
      sP[wave][(v + 8 * hi) * BKP + 16 + l16] = (__bf16)p1;
    }
    __syncthreads();

    // ---- O += P V : all fragments via 16B vector LDS loads ----
    const __bf16* pp = sP[wave] + l16 * BKP + hi * 8;
    v16bf pa = SHUF16(*(const v8bf*)pp, *(const v8bf*)(pp + 16));
#pragma unroll
    for (int f = 0; f < 4; f++) {
      const __bf16* vp = sVt + (f * 16 + l16) * BKP + hi * 8;
      v16bf bv = SHUF16(*(const v8bf*)vp, *(const v8bf*)(vp + 16));
      o[f] = __builtin_amdgcn_wmma_f32_16x16x32_bf16(
          false, pa, false, bv, (short)0, o[f], false, false);
    }
    __syncthreads();
  }

#pragma unroll
  for (int f = 0; f < 4; f++)
#pragma unroll
    for (int v = 0; v < 8; v++) {
      int row = r0 + v + 8 * hi;
      int col = h * 64 + f * 16 + l16;
      O[(size_t)row * D + col] = (__bf16)(o[f][v] / rowl[v]);
    }
}

// ---------------------------------------------------------------------------
// LayerNorm over last dim; writes f32 and optional bf16 copy.
// ---------------------------------------------------------------------------
__global__ __launch_bounds__(256)
void ln_kernel(const float* __restrict__ in, const float* __restrict__ g,
               const float* __restrict__ b, float* __restrict__ outF,
               __bf16* __restrict__ outB, int Dd)
{
  __shared__ float red0[8], red1[8];
  const int row = blockIdx.x;
  const float* p = in + (size_t)row * Dd;
  float s = 0.f, s2 = 0.f;
  for (int i = threadIdx.x; i < Dd; i += 256) { float v = p[i]; s += v; s2 += v * v; }
#pragma unroll
  for (int off = 16; off > 0; off >>= 1) {
    s  += __shfl_down(s, off);
    s2 += __shfl_down(s2, off);
  }
  int wave = threadIdx.x >> 5, lane = threadIdx.x & 31;
  if (lane == 0) { red0[wave] = s; red1[wave] = s2; }
  __syncthreads();
  float ts = 0.f, ts2 = 0.f;
#pragma unroll
  for (int i = 0; i < 8; i++) { ts += red0[i]; ts2 += red1[i]; }
  float mu = ts / Dd;
  float var = ts2 / Dd - mu * mu;
  float rstd = rsqrtf(var + 1e-5f);
  for (int i = threadIdx.x; i < Dd; i += 256) {
    float v = (p[i] - mu) * rstd * g[i] + b[i];
    if (outF) outF[(size_t)row * Dd + i] = v;
    if (outB) outB[(size_t)row * Dd + i] = (__bf16)v;
  }
}

// ---------------------------------------------------------------------------
extern "C" void kernel_launch(void* const* d_in, const int* in_sizes, int n_in,
                              void* d_out, int out_size, void* d_ws, size_t ws_size,
                              hipStream_t stream) {
  const int N = 2048, Mctx = 2048, D = 1024, H = 16, DFF = 4096, NM = 4096;

  const float* x    = (const float*)d_in[0];
  const float* ctx  = (const float*)d_in[1];
  const float* Wq   = (const float*)d_in[2];
  const float* bq   = (const float*)d_in[3];
  const float* Wkv  = (const float*)d_in[4];
  const float* bkv  = (const float*)d_in[5];
  const float* Wo   = (const float*)d_in[6];
  const float* bo   = (const float*)d_in[7];
  const float* ln1g = (const float*)d_in[8];
  const float* ln1b = (const float*)d_in[9];
  const float* W1   = (const float*)d_in[10];
  const float* bfc1 = (const float*)d_in[11];
  const float* W2   = (const float*)d_in[12];
  const float* bfc2 = (const float*)d_in[13];
  const float* ln2g = (const float*)d_in[14];
  const float* ln2b = (const float*)d_in[15];

  char* wp = (char*)d_ws;
  auto carve = [&](size_t bytes) -> void* {
    void* p = (void*)wp;
    wp += (bytes + 255) & ~(size_t)255;
    return p;
  };
  __bf16* xb    = (__bf16*)carve((size_t)N * D * 2);
  __bf16* kvin  = (__bf16*)carve((size_t)NM * D * 2);
  __bf16* wqb   = (__bf16*)carve((size_t)D * D * 2);
  __bf16* wkvb  = (__bf16*)carve((size_t)D * 2 * D * 2);
  __bf16* wob   = (__bf16*)carve((size_t)D * D * 2);
  __bf16* w1b   = (__bf16*)carve((size_t)D * DFF * 2);
  __bf16* w2b   = (__bf16*)carve((size_t)DFF * D * 2);
  __bf16* qb    = (__bf16*)carve((size_t)N * D * 2);
  __bf16* kvb   = (__bf16*)carve((size_t)NM * 2 * D * 2);
  __bf16* attnb = (__bf16*)carve((size_t)N * D * 2);
  float*  y1    = (float*)carve((size_t)N * D * 4);
  float*  x1f   = (float*)carve((size_t)N * D * 4);
  __bf16* x1b   = (__bf16*)carve((size_t)N * D * 2);
  __bf16* hb    = (__bf16*)carve((size_t)N * DFF * 2);
  float*  y2    = (float*)carve((size_t)N * D * 4);

  cvt_kernel<<<512, 256, 0, stream>>>(x, xb, N * D);
  cvt_kernel<<<512, 256, 0, stream>>>(x, kvin, N * D);
  cvt_kernel<<<512, 256, 0, stream>>>(ctx, kvin + (size_t)N * D, Mctx * D);
  cvt_kernel<<<512, 256, 0, stream>>>(Wq, wqb, D * D);
  cvt_kernel<<<512, 256, 0, stream>>>(Wkv, wkvb, D * 2 * D);
  cvt_kernel<<<512, 256, 0, stream>>>(Wo, wob, D * D);
  cvt_kernel<<<512, 256, 0, stream>>>(W1, w1b, D * DFF);
  cvt_kernel<<<512, 256, 0, stream>>>(W2, w2b, DFF * D);

  gemm_bf16<0><<<dim3(D / 128, N / 128), 256, 0, stream>>>(
      xb, wqb, bq, (const float*)nullptr, (float*)nullptr, qb, N, D, D);
  gemm_bf16<0><<<dim3(2 * D / 128, NM / 128), 256, 0, stream>>>(
      kvin, wkvb, bkv, (const float*)nullptr, (float*)nullptr, kvb, NM, 2 * D, D);
  attn_kernel<<<dim3(N / 64, H), 128, 0, stream>>>(qb, kvb, attnb);
  gemm_bf16<2><<<dim3(D / 128, N / 128), 256, 0, stream>>>(
      attnb, wob, bo, x, y1, (__bf16*)nullptr, N, D, D);
  ln_kernel<<<N, 256, 0, stream>>>(y1, ln1g, ln1b, x1f, x1b, D);
  gemm_bf16<1><<<dim3(DFF / 128, N / 128), 256, 0, stream>>>(
      x1b, w1b, bfc1, (const float*)nullptr, (float*)nullptr, hb, N, DFF, D);
  gemm_bf16<2><<<dim3(D / 128, N / 128), 256, 0, stream>>>(
      hb, w2b, bfc2, x1f, y2, (__bf16*)nullptr, N, D, DFF);
  ln_kernel<<<N, 256, 0, stream>>>(y2, ln2g, ln2b, (float*)d_out, (__bf16*)nullptr, D);
}
